// Attention_75505525064280
// MI455X (gfx1250) — matrix operand order repaired and checked by reading, not compiled, not run on hardware
//
#include <hip/hip_runtime.h>
#include <hip/hip_bf16.h>
#include <math.h>

// ---------------------------------------------------------------------------
// Multi-head attention (B=1, L=4096, C=512, H=8, D=64) for gfx1250.
// All GEMMs as A*B^T with K-major operands -> v_wmma_f32_16x16x32_f16.
// Attention: LDS-staged, double-buffered K/V tiles shared across the block.
// ---------------------------------------------------------------------------

typedef __attribute__((ext_vector_type(16))) _Float16 v16h;
typedef __attribute__((ext_vector_type(8)))  float    v8f;

#define L_SEQ 4096
#define C_DIM 512
#define HEADS 8
#define DHEAD 64
#define NQKV  1536
#define ATT_SCALE 0.125f  // 64^-0.5

union FragH { v16h h; uint4 u[2]; };

// A fragment: rows row0..row0+15, K = k0..k0+31 of a row-major (K-major) matrix.
// ISA: lanes 0-15 (M=lane) hold K = {0..7, 16..23}; lanes 16-31 hold K = {8..15, 24..31}.
__device__ __forceinline__ v16h load_frag_a_g(const _Float16* __restrict__ base,
                                              int row0, int k0, int ld) {
  const int lane = threadIdx.x & 31;
  const int row  = row0 + (lane & 15);
  const int koff = (lane >> 4) << 3;            // 0 or 8
  const _Float16* p = base + (size_t)row * ld + k0 + koff;
  FragH f;
  f.u[0] = *(const uint4*)(p);                  // K = koff .. koff+7
  f.u[1] = *(const uint4*)(p + 16);             // K = 16+koff .. 23+koff
  return f.h;
}

// B fragment for A*B^T: "columns" col0..col0+15 are rows of the K-major matrix.
// ISA: lanes 0-15 (N=lane) hold K = {0..7, 16..23}; lanes 16-31 hold K = {8..15, 24..31}.
__device__ __forceinline__ v16h load_frag_b_g(const _Float16* __restrict__ base,
                                              int col0, int k0, int ld) {
  const int lane = threadIdx.x & 31;
  const int col  = col0 + (lane & 15);
  const int koff = (lane >> 4) << 3;            // 0 or 8
  const _Float16* p = base + (size_t)col * ld + k0 + koff;
  FragH f;
  f.u[0] = *(const uint4*)(p);
  f.u[1] = *(const uint4*)(p + 16);
  return f.h;
}

// LDS-tile fragment loads (ld fixed at 64 halves). Direct indexing of the
// __shared__ arrays keeps addrspace(3) so these lower to ds_load_b128.
#define LDS_FRAG_B(dstf, bufp, col0, k0)                                  \
  do {                                                                    \
    const int _col  = (col0) + (lane & 15);                               \
    const int _koff = (lane >> 4) << 3;                                   \
    (dstf).u[0] = *(const uint4*)&(bufp)[_col * 64 + (k0) + _koff];       \
    (dstf).u[1] = *(const uint4*)&(bufp)[_col * 64 + (k0) + _koff + 16];  \
  } while (0)

#define LDS_FRAG_A(dstf, bufp, row0, k0)                                  \
  do {                                                                    \
    const int _row  = (row0) + (lane & 15);                               \
    const int _koff = (lane >> 4) << 3;                                   \
    (dstf).u[0] = *(const uint4*)&(bufp)[_row * 64 + (k0) + _koff];       \
    (dstf).u[1] = *(const uint4*)&(bufp)[_row * 64 + (k0) + _koff + 16];  \
  } while (0)

__device__ __forceinline__ v8f wmma_f16(v16h a, v16h b, v8f c) {
  return __builtin_amdgcn_wmma_f32_16x16x32_f16(false, a, false, b,
                                                (short)0, c, false, false);
}

// ---------------------------------------------------------------------------
__global__ void cvt_f32_f16(const float* __restrict__ src,
                            _Float16* __restrict__ dst, int n) {
  int i = blockIdx.x * blockDim.x + threadIdx.x;
  if (i < n) dst[i] = (_Float16)src[i];
}

// ---------------------------------------------------------------------------
// QKV projection: y[l,n] = sum_c x[l,c] * w_qkv[n,c]   (n in [0,1536))
// Epilogue scatters to Q (scaled), K (row-major [h][L][64]) and V^T ([h][64][L]).
__global__ __launch_bounds__(128) void qkv_gemm(
    const _Float16* __restrict__ xh, const _Float16* __restrict__ wh,
    _Float16* __restrict__ qws, _Float16* __restrict__ kws,
    _Float16* __restrict__ vtws) {
  const int wave = threadIdx.x >> 5;
  const int lane = threadIdx.x & 31;
  const int m0 = blockIdx.x * 128 + (wave >> 1) * 64;
  const int n0 = blockIdx.y * 128 + (wave & 1) * 64;

  v8f acc[4][4];
#pragma unroll
  for (int i = 0; i < 4; i++)
#pragma unroll
    for (int j = 0; j < 4; j++) acc[i][j] = (v8f){0, 0, 0, 0, 0, 0, 0, 0};

  for (int k = 0; k < C_DIM; k += 32) {
    v16h a[4], b[4];
#pragma unroll
    for (int i = 0; i < 4; i++) a[i] = load_frag_a_g(xh, m0 + 16 * i, k, C_DIM);
#pragma unroll
    for (int j = 0; j < 4; j++) b[j] = load_frag_b_g(wh, n0 + 16 * j, k, C_DIM);
#pragma unroll
    for (int i = 0; i < 4; i++)
#pragma unroll
      for (int j = 0; j < 4; j++) acc[i][j] = wmma_f16(a[i], b[j], acc[i][j]);
  }

#pragma unroll
  for (int i = 0; i < 4; i++) {
#pragma unroll
    for (int j = 0; j < 4; j++) {
      const int n  = n0 + 16 * j + (lane & 15);
      const int rr = m0 + 16 * i + ((lane >> 4) << 3);
#pragma unroll
      for (int r = 0; r < 8; r++) {
        const float y  = acc[i][j][r];
        const int row  = rr + r;
        if (n < 512) {                                   // Q, fold in scale
          qws[((size_t)(n >> 6) * L_SEQ + row) * DHEAD + (n & 63)] =
              (_Float16)(y * ATT_SCALE);
        } else if (n < 1024) {                           // K row-major
          const int nn = n - 512;
          kws[((size_t)(nn >> 6) * L_SEQ + row) * DHEAD + (nn & 63)] =
              (_Float16)y;
        } else {                                         // V transposed [h][d][L]
          const int nn = n - 1024;
          vtws[((size_t)(nn >> 6) * DHEAD + (nn & 63)) * (size_t)L_SEQ + row] =
              (_Float16)y;
        }
      }
    }
  }
}

// ---------------------------------------------------------------------------
// Flash attention. Block = (128 Q rows, 1 head); wave w owns 32 Q rows.
// K/V^T 64-wide tiles staged in LDS (double-buffered), shared by all 4 waves.
__global__ __launch_bounds__(128) void attn_kernel(
    const _Float16* __restrict__ qws, const _Float16* __restrict__ kws,
    const _Float16* __restrict__ vtws, _Float16* __restrict__ ows) {
  __shared__ __align__(16) _Float16 kbuf[2][64 * 64];   // K tile, rows j-rel
  __shared__ __align__(16) _Float16 vbuf[2][64 * 64];   // V^T tile, rows d
  __shared__ __align__(16) _Float16 pbuf[4][32 * 64];   // per-wave P tile

  const int tid  = threadIdx.x;
  const int wave = tid >> 5;
  const int lane = tid & 31;
  const int h    = blockIdx.y;
  const int q0   = blockIdx.x * 128 + wave * 32;

  const _Float16* qh = qws  + (size_t)h * L_SEQ * DHEAD;
  const _Float16* kh = kws  + (size_t)h * L_SEQ * DHEAD;
  const _Float16* vh = vtws + (size_t)h * DHEAD * L_SEQ;

  v8f oacc[2][4];
#pragma unroll
  for (int mi = 0; mi < 2; mi++)
#pragma unroll
    for (int t = 0; t < 4; t++) oacc[mi][t] = (v8f){0, 0, 0, 0, 0, 0, 0, 0};
  float mrow[2][8], lrow[2][8];
#pragma unroll
  for (int mi = 0; mi < 2; mi++)
#pragma unroll
    for (int r = 0; r < 8; r++) { mrow[mi][r] = -INFINITY; lrow[mi][r] = 0.0f; }

  // Q fragments are j-invariant: hoist. [m-tile][k-chunk]
  v16h aq[2][2];
#pragma unroll
  for (int mi = 0; mi < 2; mi++)
#pragma unroll
    for (int kc = 0; kc < 2; kc++)
      aq[mi][kc] = load_frag_a_g(qh, q0 + 16 * mi, 32 * kc, DHEAD);

  const int rbase = (lane >> 4) << 3;

  // ---- stage tile 0 (8KB K + 8KB V^T; 4x b128 per thread per buffer) ----
#pragma unroll
  for (int i = 0; i < 4; i++) {
    const int idx = tid + 128 * i, row = idx >> 3, c = (idx & 7) * 8;
    *(uint4*)&kbuf[0][row * 64 + c] = *(const uint4*)&kh[(size_t)row * DHEAD + c];
    *(uint4*)&vbuf[0][row * 64 + c] = *(const uint4*)&vh[(size_t)row * L_SEQ + c];
  }
  __syncthreads();

  for (int jt = 0; jt < L_SEQ / 64; ++jt) {
    const int cur = jt & 1, nxt = cur ^ 1;
    const int jn  = (jt + 1) * 64;
    const bool have_next = (jn < L_SEQ);

    // issue next tile's global loads early; they overlap the WMMA work below
    uint4 kreg[4], vreg[4];
    if (have_next) {
#pragma unroll
      for (int i = 0; i < 4; i++) {
        const int idx = tid + 128 * i, row = idx >> 3, c = (idx & 7) * 8;
        kreg[i] = *(const uint4*)&kh[(size_t)(jn + row) * DHEAD + c];
        vreg[i] = *(const uint4*)&vh[(size_t)row * L_SEQ + jn + c];
      }
    }
    // prefetch two tiles ahead into cache (global_prefetch_b8)
    const int jp = (jt + 2) * 64;
    if (jp < L_SEQ) {
      if (tid < 64) __builtin_prefetch(&kh[(size_t)(jp + tid) * DHEAD], 0, 1);
      else __builtin_prefetch(&vh[(size_t)(tid - 64) * L_SEQ + jp], 0, 1);
    }

    // ---- S = Q * K^T  (32 x 64 per wave) ----
    v8f sacc[2][4];
#pragma unroll
    for (int mi = 0; mi < 2; mi++)
#pragma unroll
      for (int t = 0; t < 4; t++) sacc[mi][t] = (v8f){0, 0, 0, 0, 0, 0, 0, 0};
#pragma unroll
    for (int kc = 0; kc < 2; kc++) {
      FragH bk[4];
#pragma unroll
      for (int t = 0; t < 4; t++) LDS_FRAG_B(bk[t], kbuf[cur], 16 * t, 32 * kc);
#pragma unroll
      for (int mi = 0; mi < 2; mi++)
#pragma unroll
        for (int t = 0; t < 4; t++)
          sacc[mi][t] = wmma_f16(aq[mi][kc], bk[t].h, sacc[mi][t]);
    }

    // ---- online softmax per 16-row m-tile ----
#pragma unroll
    for (int mi = 0; mi < 2; mi++) {
      float mx[8];
#pragma unroll
      for (int r = 0; r < 8; r++) {
        mx[r] = sacc[mi][0][r];
#pragma unroll
        for (int t = 1; t < 4; t++) mx[r] = fmaxf(mx[r], sacc[mi][t][r]);
      }
#pragma unroll
      for (int off = 8; off >= 1; off >>= 1)
#pragma unroll
        for (int r = 0; r < 8; r++)
          mx[r] = fmaxf(mx[r], __shfl_xor(mx[r], off, 32));

      float mnew[8], corr[8], rs[8];
#pragma unroll
      for (int r = 0; r < 8; r++) {
        mnew[r] = fmaxf(mrow[mi][r], mx[r]);
        corr[r] = __expf(mrow[mi][r] - mnew[r]);
        mrow[mi][r] = mnew[r];
        rs[r] = 0.0f;
      }
#pragma unroll
      for (int t = 0; t < 4; t++)
#pragma unroll
        for (int r = 0; r < 8; r++) {
          float p = __expf(sacc[mi][t][r] - mnew[r]);
          rs[r] += p;
          pbuf[wave][(mi * 16 + rbase + r) * 64 + 16 * t + (lane & 15)] =
              (_Float16)p;
        }
#pragma unroll
      for (int off = 8; off >= 1; off >>= 1)
#pragma unroll
        for (int r = 0; r < 8; r++) rs[r] += __shfl_xor(rs[r], off, 32);
#pragma unroll
      for (int r = 0; r < 8; r++)
        lrow[mi][r] = lrow[mi][r] * corr[r] + rs[r];
#pragma unroll
      for (int t = 0; t < 4; t++)
#pragma unroll
        for (int r = 0; r < 8; r++) oacc[mi][t][r] *= corr[r];
    }

    // cross-lane LDS store -> load handoff within the wave
    asm volatile("s_wait_dscnt 0" ::: "memory");

    // ---- O += P * V  (P from LDS as A-fragments, V^T rows from LDS as B) ----
#pragma unroll
    for (int j2 = 0; j2 < 64; j2 += 32) {
      FragH ap[2];
#pragma unroll
      for (int mi = 0; mi < 2; mi++) LDS_FRAG_A(ap[mi], pbuf[wave], 16 * mi, j2);
#pragma unroll
      for (int t = 0; t < 4; t++) {
        FragH bv;
        LDS_FRAG_B(bv, vbuf[cur], 16 * t, j2);
#pragma unroll
        for (int mi = 0; mi < 2; mi++)
          oacc[mi][t] = wmma_f16(ap[mi].h, bv.h, oacc[mi][t]);
      }
    }

    // commit the prefetched next tile into the spare LDS buffer
    if (have_next) {
#pragma unroll
      for (int i = 0; i < 4; i++) {
        const int idx = tid + 128 * i, row = idx >> 3, c = (idx & 7) * 8;
        *(uint4*)&kbuf[nxt][row * 64 + c] = kreg[i];
        *(uint4*)&vbuf[nxt][row * 64 + c] = vreg[i];
      }
    }
    __syncthreads();
  }

  // ---- normalize and store merged heads: ows[L][512] f16 ----
#pragma unroll
  for (int mi = 0; mi < 2; mi++) {
    float rinv[8];
#pragma unroll
    for (int r = 0; r < 8; r++) rinv[r] = 1.0f / lrow[mi][r];
#pragma unroll
    for (int t = 0; t < 4; t++) {
      const int col = h * DHEAD + 16 * t + (lane & 15);
#pragma unroll
      for (int r = 0; r < 8; r++) {
        const int row = q0 + mi * 16 + rbase + r;
        ows[(size_t)row * (HEADS * DHEAD) + col] =
            (_Float16)(oacc[mi][t][r] * rinv[r]);
      }
    }
  }
}

// ---------------------------------------------------------------------------
// Output projection: out[l,c] = sum_n O[l,n] * w_out[c,n] + b_out[c], fp32 out.
__global__ __launch_bounds__(128) void out_gemm(
    const _Float16* __restrict__ oh, const _Float16* __restrict__ wh,
    const float* __restrict__ bias, float* __restrict__ out) {
  const int wave = threadIdx.x >> 5;
  const int lane = threadIdx.x & 31;
  const int m0 = blockIdx.x * 128 + (wave >> 1) * 64;
  const int n0 = blockIdx.y * 128 + (wave & 1) * 64;

  v8f acc[4][4];
#pragma unroll
  for (int i = 0; i < 4; i++)
#pragma unroll
    for (int j = 0; j < 4; j++) acc[i][j] = (v8f){0, 0, 0, 0, 0, 0, 0, 0};

  for (int k = 0; k < C_DIM; k += 32) {
    v16h a[4], b[4];
#pragma unroll
    for (int i = 0; i < 4; i++) a[i] = load_frag_a_g(oh, m0 + 16 * i, k, C_DIM);
#pragma unroll
    for (int j = 0; j < 4; j++) b[j] = load_frag_b_g(wh, n0 + 16 * j, k, C_DIM);
#pragma unroll
    for (int i = 0; i < 4; i++)
#pragma unroll
      for (int j = 0; j < 4; j++) acc[i][j] = wmma_f16(a[i], b[j], acc[i][j]);
  }

#pragma unroll
  for (int i = 0; i < 4; i++) {
#pragma unroll
    for (int j = 0; j < 4; j++) {
      const int n  = n0 + 16 * j + (lane & 15);
      const int rr = m0 + 16 * i + ((lane >> 4) << 3);
      const float bv = bias[n];
#pragma unroll
      for (int r = 0; r < 8; r++)
        out[(size_t)(rr + r) * C_DIM + n] = acc[i][j][r] + bv;
    }
  }
}

// ---------------------------------------------------------------------------
extern "C" void kernel_launch(void* const* d_in, const int* in_sizes, int n_in,
                              void* d_out, int out_size, void* d_ws,
                              size_t ws_size, hipStream_t stream) {
  const float* x     = (const float*)d_in[0];  // [1,4096,512]
  const float* w_qkv = (const float*)d_in[1];  // [1536,512]
  const float* w_out = (const float*)d_in[2];  // [512,512]
  const float* b_out = (const float*)d_in[3];  // [512]
  float* out = (float*)d_out;

  // f16 workspace layout (elements)
  _Float16* ws    = (_Float16*)d_ws;
  _Float16* xh    = ws;                           // 4096*512
  _Float16* wqkvh = xh + (size_t)L_SEQ * C_DIM;   // 1536*512
  _Float16* wouth = wqkvh + (size_t)NQKV * C_DIM; //  512*512
  _Float16* qws   = wouth + (size_t)C_DIM * C_DIM;
  _Float16* kws   = qws + (size_t)HEADS * L_SEQ * DHEAD;
  _Float16* vtws  = kws + (size_t)HEADS * L_SEQ * DHEAD;
  _Float16* ows   = vtws + (size_t)HEADS * L_SEQ * DHEAD;

  const int nx = L_SEQ * C_DIM, nw = NQKV * C_DIM, nwo = C_DIM * C_DIM;
  cvt_f32_f16<<<(nx + 255) / 256, 256, 0, stream>>>(x, xh, nx);
  cvt_f32_f16<<<(nw + 255) / 256, 256, 0, stream>>>(w_qkv, wqkvh, nw);
  cvt_f32_f16<<<(nwo + 255) / 256, 256, 0, stream>>>(w_out, wouth, nwo);

  qkv_gemm<<<dim3(L_SEQ / 128, NQKV / 128), 128, 0, stream>>>(xh, wqkvh, qws,
                                                              kws, vtws);
  attn_kernel<<<dim3(L_SEQ / 128, HEADS), 128, 0, stream>>>(qws, kws, vtws,
                                                            ows);
  out_gemm<<<dim3(L_SEQ / 128, C_DIM / 128), 128, 0, stream>>>(ows, wouth,
                                                               b_out, out);
}
